// TransformerRecSys_58806692217038
// MI455X (gfx1250) — matrix operand order repaired and checked by reading, not compiled, not run on hardware
//
#include <hip/hip_runtime.h>
#include <stdint.h>

typedef __attribute__((ext_vector_type(16))) _Float16 v16h;
typedef __attribute__((ext_vector_type(8)))  float    v8f;
typedef uint32_t u32x4 __attribute__((ext_vector_type(4)));
typedef int      i32x4 __attribute__((ext_vector_type(4)));
typedef int      i32x8 __attribute__((ext_vector_type(8)));

#define XSH     72      // x-buffer row stride in halves (144B rows, 16B aligned)
#define WGS     64      // samples per workgroup
#define THREADS 256
#define EPSLN   1e-5f

// ---- packed-weight dword offsets inside d_ws ----
#define OFF_UPW   0
#define OFF_IPW   4096
#define OFF_INW   8192    // + l*6144
#define OFF_OUTW  20480   // + l*2048
#define OFF_FF1   24576   // + l*8192
#define OFF_FF2   40960   // + l*8192
#define OFF_H1    57344

// ---- LDS weight buffer holds d_ws dwords [8192, 61440) ----
#define LW_INW(l)  ((l)*6144)
#define LW_OUTW(l) (12288 + (l)*2048)
#define LW_FF1(l)  (16384 + (l)*8192)
#define LW_FF2(l)  (32768 + (l)*8192)
#define LW_H1      49152
#define LW_TOTAL   53248

// ---- LDS param (f32) offsets ----
#define P_UPB 0
#define P_IPB 64
#define P_PE  128
#define P_LBASE 256
#define P_LSZ 832
#define P_H1B 1920
#define P_H2W 1984
#define P_H2B 2048
#define P_TOTAL 2052

#define LDS_FENCE() asm volatile("" ::: "memory")

#if defined(__has_builtin)
#if __has_builtin(__builtin_amdgcn_tensor_load_to_lds) && \
    __has_builtin(__builtin_amdgcn_s_wait_tensorcnt)
#define HAVE_TDM 1
#endif
#endif
#ifndef HAVE_TDM
#warning "CDNA5 TDM builtins not available: falling back to manual LDS staging"
#endif

union AF { v16h h; uint32_t u[8]; };

__device__ __forceinline__ uint32_t pk2f(float a, float b){
  union { _Float16 h[2]; uint32_t u; } t;
  t.h[0]=(_Float16)a; t.h[1]=(_Float16)b; return t.u;
}
__device__ __forceinline__ float2 up2(uint32_t x){
  union { uint32_t u; _Float16 h[2]; } t; t.u=x;
  return make_float2((float)t.h[0], (float)t.h[1]);
}

__device__ __forceinline__ v8f wmma16(const AF& a, const AF& b, v8f c){
  return __builtin_amdgcn_wmma_f32_16x16x32_f16(false, a.h, false, b.h,
                                                (short)0, c, false, false);
}

// B fragment from packed buffer laid out [nt][kt][v][lane] (dwords)
__device__ __forceinline__ AF load_bfrag(const uint32_t* buf, int KT, int kt, int nt, int lane){
  AF f;
  const uint32_t* p = buf + ((nt*KT + kt) << 8) + lane;
#pragma unroll
  for (int v=0; v<8; v++) f.u[v] = p[v*32];
  return f;
}

// A fragment (16x32 f16) from an f16 buffer: two b128 loads per lane.
__device__ __forceinline__ AF load_afrag16(const _Float16* buf, int strideH, int row0,
                                           int kbase, int lane){
  AF f; const int m = lane & 15, half = lane >> 4;
  const _Float16* base = buf + (row0+m)*strideH + kbase + half*8;
  uint4 lo = *(const uint4*)base;
  uint4 hi = *(const uint4*)(base + 16);
  f.u[0]=lo.x; f.u[1]=lo.y; f.u[2]=lo.z; f.u[3]=lo.w;
  f.u[4]=hi.x; f.u[5]=hi.y; f.u[6]=hi.z; f.u[7]=hi.w;
  return f;
}

// A fragment gathered from fp32 row-major global buffer (used once, for features)
__device__ __forceinline__ AF load_afrag_f32(const float* buf, int stride, int row0,
                                             int kbase, int lane){
  AF f; const int m = lane & 15, half = lane >> 4;
#pragma unroll
  for (int v=0; v<8; v++){
    int k = kbase + ((v>>2)<<4) + (half<<3) + ((v&3)<<1);
    float2 t = *(const float2*)(buf + (row0+m)*stride + k);
    f.u[v] = pk2f(t.x, t.y);
  }
  return f;
}

// LayerNorm over 64 cols held as 4 C-tiles; reduce within 16-lane halves; store f16.
__device__ __forceinline__ void ln_store(v8f* c, const float* g, const float* b,
                                         _Float16* xb, int row0, int lane){
  const int half = lane >> 4, n = lane & 15;
#pragma unroll
  for (int r=0;r<8;r++){
    float s1 = c[0][r]+c[1][r]+c[2][r]+c[3][r];
    float s2 = c[0][r]*c[0][r]+c[1][r]*c[1][r]+c[2][r]*c[2][r]+c[3][r]*c[3][r];
#pragma unroll
    for (int msk=1; msk<16; msk<<=1){
      s1 += __shfl_xor(s1, msk, 32);
      s2 += __shfl_xor(s2, msk, 32);
    }
    float mean = s1*(1.0f/64.0f);
    float var  = s2*(1.0f/64.0f) - mean*mean;
    float rstd = rsqrtf(var + EPSLN);
    int row = row0 + r + 8*half;
#pragma unroll
    for (int nt=0;nt<4;nt++){
      int col = nt*16 + n;
      xb[row*XSH + col] = (_Float16)((c[nt][r]-mean)*rstd*g[col] + b[col]);
    }
  }
}

// -------- weight repack: src W[N][K] fp32 row-major -> packed f16 B-fragments --------
__global__ void pack_weight(const float* __restrict__ src, uint32_t* __restrict__ dst,
                            int K, int N){
  int gid = blockIdx.x*blockDim.x + threadIdx.x;
  int total = (K*N) >> 1;
  if (gid >= total) return;
  int lane = gid & 31;
  int v    = (gid >> 5) & 7;
  int tile = gid >> 8;
  int KT = K >> 5;
  int kt = tile % KT;
  int nt = tile / KT;
  int n  = nt*16 + (lane & 15);
  int k0 = kt*32 + ((v>>2)<<4) + ((lane>>4)<<3) + ((v&3)<<1);
  dst[gid] = pk2f(src[n*K + k0], src[n*K + k0 + 1]);
}

// ------------------------------ fused forward ------------------------------
__global__ __launch_bounds__(THREADS, 1)
void recsys_fused(const int* __restrict__ uidx, const int* __restrict__ iidx,
                  const float* __restrict__ ufeat, const float* __restrict__ ifeat,
                  const float* __restrict__ utab, const float* __restrict__ itab,
                  const float* __restrict__ upb, const float* __restrict__ ipb,
                  const float* __restrict__ pe,
                  const float* __restrict__ in_b, const float* __restrict__ out_b,
                  const float* __restrict__ ln1g, const float* __restrict__ ln1b,
                  const float* __restrict__ ff1b, const float* __restrict__ ff2b,
                  const float* __restrict__ ln2g, const float* __restrict__ ln2b,
                  const float* __restrict__ h1b, const float* __restrict__ h2w,
                  const float* __restrict__ h2b,
                  const uint32_t* __restrict__ wpk,
                  float* __restrict__ out){
  __shared__ __align__(16) uint32_t s_w[LW_TOTAL];     // 208 KB packed f16 weights
  __shared__ __align__(16) _Float16 s_x[128*XSH];      //  18 KB activations (f16)
  __shared__ __align__(16) _Float16 s_scr[8*16*192];   //  48 KB per-wave scratch
  __shared__ __align__(16) float    s_p[P_TOTAL];      //   8 KB biases / ln / pe

  const int tid  = threadIdx.x;
  const int lane = tid & 31;
  const int wave = tid >> 5;
  const int s_base = blockIdx.x * WGS;

  // ---- stage transformer weights global -> LDS via Tensor Data Mover ----
#ifdef HAVE_TDM
  if (wave == 0) {
    uint64_t ga = (uint64_t)(uintptr_t)(wpk + OFF_INW);
    uint32_t ldsa = (uint32_t)(uintptr_t)(&s_w[0]);   // low 32 bits of generic = LDS offset
    const uint32_t n8 = LW_TOTAL/2;                   // transfer size in 8-byte units
    u32x4 g0;
    g0[0] = 1u;                                       // count=1, user descriptor
    g0[1] = ldsa;                                     // lds_addr (bytes)
    g0[2] = (uint32_t)(ga & 0xffffffffu);             // global_addr[31:0]
    g0[3] = (uint32_t)((ga >> 32) & 0x01ffffffu) | 0x80000000u; // addr[56:32] | type=2
    i32x8 g1;
    g1[0] = (int)(3u << 16);                          // data_size = 8B, mask=0
    g1[1] = (int)((n8 & 0xffffu) << 16);              // tensor_dim0[15:0]
    g1[2] = (int)((n8 >> 16) & 0xffffu) | (1 << 16);  // tensor_dim0[31:16] | tensor_dim1=1
    g1[3] = (int)(n8 << 16);                          // tile_dim0
    g1[4] = 0;                                        // tile_dim1/2 unused
    g1[5] = (int)n8;                                  // tensor_dim0_stride[31:0]
    g1[6] = 0;
    g1[7] = 0;
    i32x4 z4 = {0,0,0,0};
#if __clang_major__ >= 23
    i32x8 z8 = {0,0,0,0,0,0,0,0};
    __builtin_amdgcn_tensor_load_to_lds(g0, g1, z4, z4, z8, 0);
#else
    __builtin_amdgcn_tensor_load_to_lds(g0, g1, z4, z4, 0);
#endif
    __builtin_amdgcn_s_wait_tensorcnt((short)0);
  }
#else
  {
    const uint4* src = (const uint4*)(wpk + OFF_INW);
    uint4* dst = (uint4*)s_w;
    for (int i = tid; i < LW_TOTAL/4; i += THREADS) dst[i] = src[i];
  }
#endif
  // ---- stage small params ----
  for (int i=tid;i<64;i+=THREADS){
    s_p[P_UPB+i]=upb[i]; s_p[P_IPB+i]=ipb[i];
    s_p[P_H1B+i]=h1b[i]; s_p[P_H2W+i]=h2w[i];
  }
  for (int i=tid;i<128;i+=THREADS) s_p[P_PE+i]=pe[i];
#pragma unroll
  for (int l=0;l<2;l++){
    const int pb = P_LBASE + l*P_LSZ;
    for (int i=tid;i<192;i+=THREADS) s_p[pb+i]=in_b[l*192+i];
    for (int i=tid;i<64;i+=THREADS){
      s_p[pb+192+i]=out_b[l*64+i];
      s_p[pb+256+i]=ln1g[l*64+i];
      s_p[pb+320+i]=ln1b[l*64+i];
      s_p[pb+640+i]=ff2b[l*64+i];
      s_p[pb+704+i]=ln2g[l*64+i];
      s_p[pb+768+i]=ln2b[l*64+i];
    }
    for (int i=tid;i<256;i+=THREADS) s_p[pb+384+i]=ff1b[l*256+i];
  }
  if (tid==0) s_p[P_H2B]=h2b[0];
  __syncthreads();

  // =============== stage 1: embed + input projection (GEMM K=128) ===============
  {
    const bool isUser = (wave < 4);
    const int tile = wave & 3;
    const int s0 = s_base + tile*16;
    const float* feat = isUser ? ufeat : ifeat;
    const float* tab  = isUser ? utab  : itab;
    const int*   idx  = isUser ? uidx  : iidx;
    const float* bias = &s_p[isUser ? P_UPB : P_IPB];
    const float* pev  = &s_p[P_PE + (isUser ? 0 : 64)];
    const uint32_t* wb = wpk + (isUser ? OFF_UPW : OFF_IPW);  // L2-resident
    const int n = lane & 15, half = lane >> 4;
    AF a[4];
#pragma unroll
    for (int kt=0;kt<4;kt++) a[kt] = load_afrag_f32(feat, 128, s0, kt*32, lane);
#pragma unroll
    for (int nt=0;nt<4;nt++){
      v8f c = {0.f,0.f,0.f,0.f,0.f,0.f,0.f,0.f};
#pragma unroll
      for (int kt=0;kt<4;kt++) c = wmma16(a[kt], load_bfrag(wb, 4, kt, nt, lane), c);
      float bp = bias[nt*16+n] + pev[nt*16+n];
#pragma unroll
      for (int r=0;r<8;r++){
        int s = s0 + r + 8*half;
        int col = nt*16 + n;
        float val = c[r] + bp + tab[(size_t)idx[s]*64 + col];
        int xrow = (s - s_base)*2 + (isUser ? 0 : 1);
        s_x[xrow*XSH + col] = (_Float16)val;
      }
    }
  }
  __syncthreads();

  // =============== stage 2: transformer layers (wave-local, 16 rows) ===============
  const int row0 = wave * 16;
  uint32_t* scr   = (uint32_t*)s_scr + wave*1536;    // per-wave [16][192] f16
  _Float16* scr16 = s_scr + wave*3072;

  for (int l=0;l<2;l++){
    const float* pl = &s_p[P_LBASE + l*P_LSZ];
    const uint32_t* w_in  = s_w + LW_INW(l);
    const uint32_t* w_out = s_w + LW_OUTW(l);
    const uint32_t* w_f1  = s_w + LW_FF1(l);
    const uint32_t* w_f2  = s_w + LW_FF2(l);
    const int n = lane & 15, half = lane >> 4;

    // ---- (a) QKV: [16,64] x [64,192] ----
    {
      AF a0 = load_afrag16(s_x, XSH, row0, 0,  lane);
      AF a1 = load_afrag16(s_x, XSH, row0, 32, lane);
#pragma unroll
      for (int nt=0;nt<12;nt++){
        v8f c = {0.f,0.f,0.f,0.f,0.f,0.f,0.f,0.f};
        c = wmma16(a0, load_bfrag(w_in, 2, 0, nt, lane), c);
        c = wmma16(a1, load_bfrag(w_in, 2, 1, nt, lane), c);
        float bv = pl[nt*16+n];
#pragma unroll
        for (int r=0;r<8;r++)
          scr16[(r + 8*half)*192 + nt*16 + n] = (_Float16)(c[r] + bv);
      }
    }
    LDS_FENCE();

    // ---- (b) attention: lane = (local sample, head); seq len 2 ----
    {
      const int sl2 = (lane >> 2) * 2;
      const int hh  = lane & 3;
      const uint32_t* r0p = scr + sl2*96;
      const uint32_t* r1p = scr + (sl2+1)*96;
      float q0[16],q1[16],k0v[16],k1v[16],v0[16],v1[16];
#pragma unroll
      for (int i=0;i<8;i++){
        float2 t;
        t = up2(r0p[hh*8+i]);     q0[2*i]=t.x;  q0[2*i+1]=t.y;
        t = up2(r1p[hh*8+i]);     q1[2*i]=t.x;  q1[2*i+1]=t.y;
        t = up2(r0p[32+hh*8+i]);  k0v[2*i]=t.x; k0v[2*i+1]=t.y;
        t = up2(r1p[32+hh*8+i]);  k1v[2*i]=t.x; k1v[2*i+1]=t.y;
        t = up2(r0p[64+hh*8+i]);  v0[2*i]=t.x;  v0[2*i+1]=t.y;
        t = up2(r1p[64+hh*8+i]);  v1[2*i]=t.x;  v1[2*i+1]=t.y;
      }
      float s00=0,s01=0,s10=0,s11=0;
#pragma unroll
      for (int i=0;i<16;i++){
        s00 += q0[i]*k0v[i]; s01 += q0[i]*k1v[i];
        s10 += q1[i]*k0v[i]; s11 += q1[i]*k1v[i];
      }
      s00*=0.25f; s01*=0.25f; s10*=0.25f; s11*=0.25f;   // 1/sqrt(16)
      float m0=fmaxf(s00,s01), e0=__expf(s00-m0), e1=__expf(s01-m0), i0=1.f/(e0+e1);
      float a00=e0*i0, a01=e1*i0;
      float m1=fmaxf(s10,s11), f0=__expf(s10-m1), f1=__expf(s11-m1), i1=1.f/(f0+f1);
      float a10=f0*i1, a11=f1*i1;
      uint32_t* w0p = scr + sl2*96 + hh*8;        // ctx overwrites q region
      uint32_t* w1p = scr + (sl2+1)*96 + hh*8;
#pragma unroll
      for (int i=0;i<8;i++){
        w0p[i] = pk2f(a00*v0[2*i]+a01*v1[2*i], a00*v0[2*i+1]+a01*v1[2*i+1]);
        w1p[i] = pk2f(a10*v0[2*i]+a11*v1[2*i], a10*v0[2*i+1]+a11*v1[2*i+1]);
      }
    }
    LDS_FENCE();

    // ---- (c) out-proj + residual + LN1 ----
    {
      AF a0 = load_afrag16(scr16, 192, 0, 0,  lane);
      AF a1 = load_afrag16(scr16, 192, 0, 32, lane);
      v8f c[4];
#pragma unroll
      for (int nt=0;nt<4;nt++){
        v8f t = {0.f,0.f,0.f,0.f,0.f,0.f,0.f,0.f};
        t = wmma16(a0, load_bfrag(w_out, 2, 0, nt, lane), t);
        t = wmma16(a1, load_bfrag(w_out, 2, 1, nt, lane), t);
        float bv = pl[192 + nt*16 + n];
#pragma unroll
        for (int r=0;r<8;r++)
          t[r] += bv + (float)s_x[(row0 + r + 8*half)*XSH + nt*16 + n];
        c[nt] = t;
      }
      ln_store(c, &pl[256], &pl[320], s_x, row0, lane);
    }
    LDS_FENCE();

    // ---- (d) FF: 64 -> 256 (relu) -> 64, K-chunked through 1KB scratch ----
    {
      AF ax0 = load_afrag16(s_x, XSH, row0, 0,  lane);
      AF ax1 = load_afrag16(s_x, XSH, row0, 32, lane);
      v8f c2[4];
#pragma unroll
      for (int nt=0;nt<4;nt++){
        v8f t = {0.f,0.f,0.f,0.f,0.f,0.f,0.f,0.f};
        c2[nt] = t;
      }
#pragma unroll
      for (int kc=0;kc<8;kc++){
#pragma unroll
        for (int j=0;j<2;j++){
          int nt = kc*2 + j;
          v8f ch = {0.f,0.f,0.f,0.f,0.f,0.f,0.f,0.f};
          ch = wmma16(ax0, load_bfrag(w_f1, 2, 0, nt, lane), ch);
          ch = wmma16(ax1, load_bfrag(w_f1, 2, 1, nt, lane), ch);
          float bv = pl[384 + nt*16 + n];
#pragma unroll
          for (int r=0;r<8;r++)
            scr16[(r + 8*half)*32 + j*16 + n] = (_Float16)fmaxf(ch[r] + bv, 0.f);
        }
        LDS_FENCE();
        AF ah = load_afrag16(scr16, 32, 0, 0, lane);
#pragma unroll
        for (int nt=0;nt<4;nt++)
          c2[nt] = wmma16(ah, load_bfrag(w_f2, 8, kc, nt, lane), c2[nt]);
        LDS_FENCE();
      }
#pragma unroll
      for (int nt=0;nt<4;nt++){
        float bv = pl[640 + nt*16 + n];
#pragma unroll
        for (int r=0;r<8;r++)
          c2[nt][r] += bv + (float)s_x[(row0 + r + 8*half)*XSH + nt*16 + n];
      }
      ln_store(c2, &pl[704], &pl[768], s_x, row0, lane);
    }
    LDS_FENCE();
  }
  __syncthreads();

  // =============== stage 3: head (waves 0..3, 16 samples each) ===============
  if (wave < 4){
    const int s0l = wave*16;
    const int n = lane & 15, half = lane >> 4;
    const uint32_t* wh = s_w + LW_H1;
    AF a[4];
#pragma unroll
    for (int kt=0;kt<4;kt++){
      // k-chunk kt: token row = kt>>1 (k<64 vs k>=64), col base = (kt&1)*32
      a[kt] = load_afrag16(s_x + (2*s0l + (kt>>1))*XSH, 2*XSH, 0, (kt&1)*32, lane);
    }
    v8f c[4];
#pragma unroll
    for (int nt=0;nt<4;nt++){
      v8f t = {0.f,0.f,0.f,0.f,0.f,0.f,0.f,0.f};
#pragma unroll
      for (int kt=0;kt<4;kt++) t = wmma16(a[kt], load_bfrag(wh, 4, kt, nt, lane), t);
      c[nt] = t;
    }
    float w2[4], b1[4];
#pragma unroll
    for (int nt=0;nt<4;nt++){
      w2[nt] = s_p[P_H2W + nt*16 + n];
      b1[nt] = s_p[P_H1B + nt*16 + n];
    }
#pragma unroll
    for (int r=0;r<8;r++){
      float t = 0.f;
#pragma unroll
      for (int nt=0;nt<4;nt++) t += fmaxf(c[nt][r] + b1[nt], 0.f) * w2[nt];
#pragma unroll
      for (int msk=1; msk<16; msk<<=1) t += __shfl_xor(t, msk, 32);
      if (n == 0)
        out[s_base + s0l + r + 8*half] = t + s_p[P_H2B];
    }
  }
}

// ------------------------------ host launch ------------------------------
extern "C" void kernel_launch(void* const* d_in, const int* in_sizes, int n_in,
                              void* d_out, int out_size, void* d_ws, size_t ws_size,
                              hipStream_t stream) {
  const int*   uidx  = (const int*)  d_in[0];
  const int*   iidx  = (const int*)  d_in[1];
  const float* ufeat = (const float*)d_in[2];
  const float* ifeat = (const float*)d_in[3];
  const float* utab  = (const float*)d_in[4];
  const float* itab  = (const float*)d_in[5];
  const float* upw   = (const float*)d_in[6];
  const float* upb   = (const float*)d_in[7];
  const float* ipw   = (const float*)d_in[8];
  const float* ipb   = (const float*)d_in[9];
  const float* pe    = (const float*)d_in[10];
  const float* in_w  = (const float*)d_in[11];
  const float* in_b  = (const float*)d_in[12];
  const float* out_w = (const float*)d_in[13];
  const float* out_b = (const float*)d_in[14];
  const float* ln1g  = (const float*)d_in[15];
  const float* ln1b  = (const float*)d_in[16];
  const float* ff1w  = (const float*)d_in[17];
  const float* ff1b  = (const float*)d_in[18];
  const float* ff2w  = (const float*)d_in[19];
  const float* ff2b  = (const float*)d_in[20];
  const float* ln2g  = (const float*)d_in[21];
  const float* ln2b  = (const float*)d_in[22];
  const float* h1w   = (const float*)d_in[23];
  const float* h1b   = (const float*)d_in[24];
  const float* h2w   = (const float*)d_in[25];
  const float* h2b   = (const float*)d_in[26];
  float*    out = (float*)d_out;
  uint32_t* wpk = (uint32_t*)d_ws;
  const int B = in_sizes[0];

  auto pack = [&](const float* s, int off, int K, int N){
    int tot = (K*N) >> 1;
    pack_weight<<<dim3((tot+255)/256), dim3(256), 0, stream>>>(s, wpk+off, K, N);
  };
  pack(upw, OFF_UPW, 128, 64);
  pack(ipw, OFF_IPW, 128, 64);
  for (int l=0;l<2;l++){
    pack(in_w  + l*192*64,  OFF_INW  + l*6144, 64, 192);
    pack(out_w + l*64*64,   OFF_OUTW + l*2048, 64, 64);
    pack(ff1w  + l*256*64,  OFF_FF1  + l*8192, 64, 256);
    pack(ff2w  + l*64*256,  OFF_FF2  + l*8192, 256, 64);
  }
  pack(h1w, OFF_H1, 128, 64);

  int nb = B / WGS;
  recsys_fused<<<dim3(nb), dim3(THREADS), 0, stream>>>(
      uidx, iidx, ufeat, ifeat, utab, itab,
      upb, ipb, pe, in_b, out_b, ln1g, ln1b, ff1b, ff2b, ln2g, ln2b,
      h1b, h2w, h2b, wpk, out);
}
